// SCRNLayer_69939247448808
// MI455X (gfx1250) — matrix operand order repaired and checked by reading, not compiled, not run on hardware
//
#include <hip/hip_runtime.h>
#include <hip/hip_bf16.h>
#include <math.h>

// ---------------------------------------------------------------------------
// SCRN layer for MI455X (gfx1250), wave32 + WMMA bf16 (f32 accumulate) +
// async global->LDS tile staging (ASYNCcnt path).
//
//   B=64, T=512, IN=1024, H=1024, C=512, ALPHA=0.95
//
// Kernels:
//   cvt_bf16       : fp32 -> bf16 (x, Wx, Uc, Vh; row-major [N][K] == WMMA-B)
//   gemm_xc_async  : xc = x*Wx^T + bx, bf16 WMMA, tiles staged with
//                    global_load_async_to_lds_b128, double-buffered with
//                    s_wait_asynccnt 4 (prev tile done, next in flight).
//   gemm_xc_cvt    : fallback (fp32 load + inline convert) if ws is small.
//   scrn_scan      : 4 persistent blocks, one 16-row batch tile each;
//                    c (fp32) / h (bf16) LDS-resident across all 512 steps;
//                    48 K-iters of v_wmma_f32_16x16x32_bf16 per wave per step.
//
// Workspace (async path, bytes):
//   [0,          67108864)   xc   fp32 [32768][512]
//   [67108864,  134217728)   xbf  bf16 [32768][1024]
//   [134217728, 135266304)   WxB  bf16 [512][1024]
//   [135266304, 136314880)   UcB  bf16 [1024][512]
//   [136314880, 138412032)   VhB  bf16 [1024][1024]
// ---------------------------------------------------------------------------

typedef __attribute__((ext_vector_type(16))) __bf16 v16bf;
typedef __attribute__((ext_vector_type(8)))  __bf16 v8bf;
typedef __attribute__((ext_vector_type(8)))  float  v8f;
typedef __attribute__((ext_vector_type(4)))  float  v4f;

union Frag16 { v16bf v; v8bf h[2]; };

__device__ __forceinline__ unsigned short f2bf(float f) {
  union { float f; unsigned u; } v; v.f = f;
  unsigned r = v.u + 0x7FFFu + ((v.u >> 16) & 1u);   // round-to-nearest-even
  return (unsigned short)(r >> 16);
}

// one 16B async copy: global -> LDS (raw bytes, tracked by ASYNCcnt)
__device__ __forceinline__ void async_b128(unsigned lds_addr, const void* gaddr) {
  asm volatile("global_load_async_to_lds_b128 %0, %1, off"
               :: "v"(lds_addr), "v"((unsigned long long)(uintptr_t)gaddr)
               : "memory");
}

// ---------------------------------------------------------------------------
__global__ __launch_bounds__(256) void cvt_bf16(const float* __restrict__ s,
                                                unsigned short* __restrict__ d,
                                                int n) {
  int i = (blockIdx.x * 256 + threadIdx.x) * 4;
  if (i + 3 < n) {
    v4f v = *(const v4f*)(s + i);
    d[i + 0] = f2bf(v.x);
    d[i + 1] = f2bf(v.y);
    d[i + 2] = f2bf(v.z);
    d[i + 3] = f2bf(v.w);
  }
}

// ---------------------------------------------------------------------------
// Shared GEMM geometry: M=32768, N=512, K=1024. Block tile 128x128, K-step 32.
// 256 threads = 8 waves (2 M-slabs x 4 N-slabs); wave owns 64x32 = 8 tiles.
// LDS rows: 32 bf16 (64B payload) with 80B stride (16B-aligned, 20-bank skew).
// ---------------------------------------------------------------------------
#define G_PAD 40         // row stride in ushorts (80 B)
#define G_BUF 10240      // one 128-row tile buffer, bytes

__global__ __launch_bounds__(256) void gemm_xc_async(
    const unsigned short* __restrict__ xbf,   // [32768][1024] bf16
    const unsigned short* __restrict__ WxB,   // [512][1024]   bf16
    const float* __restrict__ bx,
    float* __restrict__ xc) {
  extern __shared__ char smem[];
  unsigned short* As = (unsigned short*)smem;             // 2 x [128][G_PAD]
  unsigned short* Bs = As + G_BUF;                        // (A0 B0 A1 B1)

  const int tid  = threadIdx.x;
  const int lane = tid & 31;
  const int wave = tid >> 5;
  const int wm   = wave >> 2;
  const int wn   = wave & 3;
  const int bM   = blockIdx.x & 255;
  const int bN   = blockIdx.x >> 8;
  const int m0   = bM * 128;
  const int n0   = bN * 128;
  const int half = lane >> 4;
  const int lr   = lane & 15;

  const unsigned ldsA = (unsigned)(uintptr_t)As;
  const unsigned ldsB = (unsigned)(uintptr_t)Bs;

  // stage one 128x32 bf16 tile pair: 2 chunks of 16B per thread per matrix
  auto stage = [&](int k0, int buf) {
    unsigned aB = ldsA + buf * (2 * G_BUF);
    unsigned bB = ldsB + buf * (2 * G_BUF);
#pragma unroll
    for (int i = 0; i < 2; ++i) {
      int ch  = tid * 2 + i;           // 0..511
      int r   = ch >> 2;               // row 0..127
      int off = (ch & 3) * 16;         // byte offset in 64B row payload
      async_b128(aB + r * 80 + off,
                 (const char*)xbf + ((size_t)(m0 + r) * 1024 + k0) * 2 + off);
      async_b128(bB + r * 80 + off,
                 (const char*)WxB + ((size_t)(n0 + r) * 1024 + k0) * 2 + off);
    }
  };

  v8f acc[4][2];
#pragma unroll
  for (int jm = 0; jm < 4; ++jm)
#pragma unroll
    for (int jn = 0; jn < 2; ++jn) acc[jm][jn] = (v8f)(0.0f);

  stage(0, 0);                                   // prologue
  for (int kk = 0; kk < 32; ++kk) {
    if (kk + 1 < 32) {
      stage((kk + 1) * 32, (kk + 1) & 1);        // overlap next tile
      asm volatile("s_wait_asynccnt 0x4" ::: "memory");  // tile kk complete
    } else {
      asm volatile("s_wait_asynccnt 0x0" ::: "memory");
    }
    __syncthreads();

    const unsigned short* Ab = As + (kk & 1) * (G_BUF);   // ushort units
    const unsigned short* Bb = Bs + (kk & 1) * (G_BUF);
    Frag16 a[4], b[2];
#pragma unroll
    for (int jm = 0; jm < 4; ++jm) {
      const unsigned short* p = Ab + (wm * 64 + jm * 16 + lr) * G_PAD;
      a[jm].h[0] = *(const v8bf*)(p + (half ? 16 : 0));   // K {0..7}|{16..23}
      a[jm].h[1] = *(const v8bf*)(p + (half ? 24 : 8));   // K {8..15}|{24..31}
    }
#pragma unroll
    for (int jn = 0; jn < 2; ++jn) {
      const unsigned short* p = Bb + (wn * 32 + jn * 16 + lr) * G_PAD + (half ? 16 : 0);
      b[jn].h[0] = *(const v8bf*)(p);
      b[jn].h[1] = *(const v8bf*)(p + 8);
    }
#pragma unroll
    for (int jm = 0; jm < 4; ++jm)
#pragma unroll
      for (int jn = 0; jn < 2; ++jn)
        acc[jm][jn] = __builtin_amdgcn_wmma_f32_16x16x32_bf16(
            false, a[jm].v, false, b[jn].v, (short)0, acc[jm][jn], false, false);
    __syncthreads();   // reads done before this buffer is overwritten
  }

#pragma unroll
  for (int jm = 0; jm < 4; ++jm) {
#pragma unroll
    for (int jn = 0; jn < 2; ++jn) {
      int colg  = n0 + wn * 32 + jn * 16 + lr;
      float bia = bx[colg];
      int mrow0 = m0 + wm * 64 + jm * 16 + (half ? 8 : 0);
#pragma unroll
      for (int r = 0; r < 8; ++r)
        xc[(size_t)(mrow0 + r) * 512 + colg] = acc[jm][jn][r] + bia;
    }
  }
}

// ---------------------------------------------------------------------------
// Fallback GEMM: fp32 loads + inline bf16 convert into LDS (no xbf needed).
// ---------------------------------------------------------------------------
__global__ __launch_bounds__(256) void gemm_xc_cvt(const float* __restrict__ x,
                                                   const float* __restrict__ Wx,
                                                   const float* __restrict__ bx,
                                                   float* __restrict__ xc) {
  extern __shared__ char smem[];
  unsigned short* As = (unsigned short*)smem;        // [128][G_PAD]
  unsigned short* Bs = As + 128 * G_PAD;

  const int tid  = threadIdx.x;
  const int lane = tid & 31;
  const int wave = tid >> 5;
  const int wm   = wave >> 2;
  const int wn   = wave & 3;
  const int bM   = blockIdx.x & 255;
  const int bN   = blockIdx.x >> 8;
  const int m0   = bM * 128;
  const int n0   = bN * 128;
  const int half = lane >> 4;
  const int lr   = lane & 15;

  v8f acc[4][2];
#pragma unroll
  for (int jm = 0; jm < 4; ++jm)
#pragma unroll
    for (int jn = 0; jn < 2; ++jn) acc[jm][jn] = (v8f)(0.0f);

  for (int k0 = 0; k0 < 1024; k0 += 32) {
    __syncthreads();
#pragma unroll
    for (int it = 0; it < 4; ++it) {
      int chunk = it * 256 + tid;
      int r  = chunk >> 3;
      int c4 = (chunk & 7) * 4;
      v4f xv = *(const v4f*)(x + (size_t)(m0 + r) * 1024 + k0 + c4);
      unsigned short* da = As + r * G_PAD + c4;
      da[0] = f2bf(xv.x); da[1] = f2bf(xv.y); da[2] = f2bf(xv.z); da[3] = f2bf(xv.w);
      v4f wv = *(const v4f*)(Wx + (size_t)(n0 + r) * 1024 + k0 + c4);
      unsigned short* db = Bs + r * G_PAD + c4;
      db[0] = f2bf(wv.x); db[1] = f2bf(wv.y); db[2] = f2bf(wv.z); db[3] = f2bf(wv.w);
    }
    __syncthreads();

    Frag16 a[4], b[2];
#pragma unroll
    for (int jm = 0; jm < 4; ++jm) {
      const unsigned short* p = As + (wm * 64 + jm * 16 + lr) * G_PAD;
      a[jm].h[0] = *(const v8bf*)(p + (half ? 16 : 0));
      a[jm].h[1] = *(const v8bf*)(p + (half ? 24 : 8));
    }
#pragma unroll
    for (int jn = 0; jn < 2; ++jn) {
      const unsigned short* p = Bs + (wn * 32 + jn * 16 + lr) * G_PAD + (half ? 16 : 0);
      b[jn].h[0] = *(const v8bf*)(p);
      b[jn].h[1] = *(const v8bf*)(p + 8);
    }
#pragma unroll
    for (int jm = 0; jm < 4; ++jm)
#pragma unroll
      for (int jn = 0; jn < 2; ++jn)
        acc[jm][jn] = __builtin_amdgcn_wmma_f32_16x16x32_bf16(
            false, a[jm].v, false, b[jn].v, (short)0, acc[jm][jn], false, false);
  }

#pragma unroll
  for (int jm = 0; jm < 4; ++jm) {
#pragma unroll
    for (int jn = 0; jn < 2; ++jn) {
      int colg  = n0 + wn * 32 + jn * 16 + lr;
      float bia = bx[colg];
      int mrow0 = m0 + wm * 64 + jm * 16 + (half ? 8 : 0);
#pragma unroll
      for (int r = 0; r < 8; ++r)
        xc[(size_t)(mrow0 + r) * 512 + colg] = acc[jm][jn][r] + bia;
    }
  }
}

// ---------------------------------------------------------------------------
// Persistent scan. 4 blocks x 256 threads; block owns batch rows [16*blk,+16).
// LDS: Cst fp32[16][512], Cbf bf16[16][S_CP], Hbf bf16[16][S_HP].
// ---------------------------------------------------------------------------
#define S_CP 520
#define S_HP 1032

__global__ __launch_bounds__(256) void scrn_scan(
    const float* __restrict__ xc,            // [32768][512]
    const float* __restrict__ h0,            // [64][1024]
    const float* __restrict__ c0,            // [64][512]
    const unsigned short* __restrict__ UcB,  // [1024][512]  bf16
    const unsigned short* __restrict__ VhB,  // [1024][1024] bf16
    float* __restrict__ out) {               // hs | h_T | c_T
  extern __shared__ char smem[];
  float*          Cst = (float*)smem;
  unsigned short* Cbf = (unsigned short*)(smem + 16 * 512 * 4);
  unsigned short* Hbf = Cbf + 16 * S_CP;

  const int tid  = threadIdx.x;
  const int lane = tid & 31;
  const int wave = tid >> 5;
  const int m0   = blockIdx.x * 16;
  const int half = lane >> 4;
  const int lr   = lane & 15;
  const int nb   = wave * 128;

  const size_t HS_ELEMS = (size_t)64 * 512 * 1024;
  const size_t HT_OFF   = HS_ELEMS;
  const size_t CT_OFF   = HS_ELEMS + (size_t)64 * 1024;

#pragma unroll
  for (int i = 0; i < 8; ++i) {
    int chunk = i * 256 + tid;
    int r = chunk >> 7, c4 = (chunk & 127) * 4;
    *(v4f*)(Cst + r * 512 + c4) = *(const v4f*)(c0 + (size_t)(m0 + r) * 512 + c4);
  }
#pragma unroll
  for (int i = 0; i < 16; ++i) {
    int chunk = i * 256 + tid;
    int r = chunk >> 8, c4 = (chunk & 255) * 4;
    v4f hv = *(const v4f*)(h0 + (size_t)(m0 + r) * 1024 + c4);
    unsigned short* d = Hbf + r * S_HP + c4;
    d[0] = f2bf(hv.x); d[1] = f2bf(hv.y); d[2] = f2bf(hv.z); d[3] = f2bf(hv.w);
  }
  __syncthreads();

  for (int t = 0; t < 512; ++t) {
#pragma unroll
    for (int i = 0; i < 8; ++i) {
      int chunk = i * 256 + tid;
      int r = chunk >> 7, c4 = (chunk & 127) * 4;
      v4f xv = *(const v4f*)(xc + ((size_t)(m0 + r) * 512 + t) * 512 + c4);
      float* cp = Cst + r * 512 + c4;
      v4f cv = *(v4f*)cp;
      cv.x = 0.05f * cv.x + 0.95f * xv.x;
      cv.y = 0.05f * cv.y + 0.95f * xv.y;
      cv.z = 0.05f * cv.z + 0.95f * xv.z;
      cv.w = 0.05f * cv.w + 0.95f * xv.w;
      *(v4f*)cp = cv;
      unsigned short* cb = Cbf + r * S_CP + c4;
      cb[0] = f2bf(cv.x); cb[1] = f2bf(cv.y); cb[2] = f2bf(cv.z); cb[3] = f2bf(cv.w);
    }
    if (t < 511)  // warm L2/WGP$ for next step's xc slice (global_prefetch_b8)
      __builtin_prefetch(xc + ((size_t)(m0 + (tid & 15)) * 512 + (t + 1)) * 512 + (tid >> 4) * 32, 0, 3);
    __syncthreads();

    v8f acc[8];
#pragma unroll
    for (int j = 0; j < 8; ++j) acc[j] = (v8f)(0.0f);

    for (int k0 = 0; k0 < 512; k0 += 32) {            // c_t * Uc^T
      Frag16 a;
      const unsigned short* ap = Cbf + lr * S_CP + k0;
      a.h[0] = *(const v8bf*)(ap + (half ? 16 : 0));
      a.h[1] = *(const v8bf*)(ap + (half ? 24 : 8));
#pragma unroll
      for (int j = 0; j < 8; ++j) {
        int n = nb + j * 16 + lr;
        const unsigned short* bp = UcB + (size_t)n * 512 + k0 + (half ? 16 : 0);
        Frag16 b;
        b.h[0] = *(const v8bf*)(bp);
        b.h[1] = *(const v8bf*)(bp + 8);
        acc[j] = __builtin_amdgcn_wmma_f32_16x16x32_bf16(
            false, a.v, false, b.v, (short)0, acc[j], false, false);
      }
    }
    for (int k0 = 0; k0 < 1024; k0 += 32) {           // h_prev * Vh^T
      Frag16 a;
      const unsigned short* ap = Hbf + lr * S_HP + k0;
      a.h[0] = *(const v8bf*)(ap + (half ? 16 : 0));
      a.h[1] = *(const v8bf*)(ap + (half ? 24 : 8));
#pragma unroll
      for (int j = 0; j < 8; ++j) {
        int n = nb + j * 16 + lr;
        const unsigned short* bp = VhB + (size_t)n * 1024 + k0 + (half ? 16 : 0);
        Frag16 b;
        b.h[0] = *(const v8bf*)(bp);
        b.h[1] = *(const v8bf*)(bp + 8);
        acc[j] = __builtin_amdgcn_wmma_f32_16x16x32_bf16(
            false, a.v, false, b.v, (short)0, acc[j], false, false);
      }
    }
    __syncthreads();

#pragma unroll
    for (int j = 0; j < 8; ++j) {
      int n   = nb + j * 16 + lr;
      int mr0 = half ? 8 : 0;
#pragma unroll
      for (int r = 0; r < 8; ++r) {
        float hv = tanhf(acc[j][r]);
        int m = mr0 + r;
        out[((size_t)(m0 + m) * 512 + t) * 1024 + n] = hv;
        Hbf[m * S_HP + n] = f2bf(hv);
        if (t == 511) out[HT_OFF + (size_t)(m0 + m) * 1024 + n] = hv;
      }
    }
    __syncthreads();
  }

#pragma unroll
  for (int i = 0; i < 8; ++i) {
    int chunk = i * 256 + tid;
    int r = chunk >> 7, c4 = (chunk & 127) * 4;
    *(v4f*)(out + CT_OFF + (size_t)(m0 + r) * 512 + c4) = *(const v4f*)(Cst + r * 512 + c4);
  }
}

// ---------------------------------------------------------------------------
extern "C" void kernel_launch(void* const* d_in, const int* in_sizes, int n_in,
                              void* d_out, int out_size, void* d_ws, size_t ws_size,
                              hipStream_t stream) {
  const float* x  = (const float*)d_in[0];   // [64,512,1024]
  const float* h0 = (const float*)d_in[1];   // [64,1024]
  const float* c0 = (const float*)d_in[2];   // [64,512]
  const float* Wx = (const float*)d_in[3];   // [512,1024]
  const float* bx = (const float*)d_in[4];   // [512]
  const float* Uc = (const float*)d_in[5];   // [1024,512]
  const float* Vh = (const float*)d_in[6];   // [1024,1024]
  float* out = (float*)d_out;

  char* ws = (char*)d_ws;
  const size_t XC_BYTES  = (size_t)32768 * 512 * 4;    //  67,108,864
  const size_t XBF_BYTES = (size_t)32768 * 1024 * 2;   //  67,108,864
  const size_t WX_BYTES  = (size_t)512 * 1024 * 2;     //   1,048,576
  const size_t UC_BYTES  = (size_t)1024 * 512 * 2;     //   1,048,576
  const size_t VH_BYTES  = (size_t)1024 * 1024 * 2;    //   2,097,152
  const size_t FULL      = XC_BYTES + XBF_BYTES + WX_BYTES + UC_BYTES + VH_BYTES;

  const bool big_ws = (ws_size >= FULL);

  float* xcb = (float*)ws;
  unsigned short *xbf, *WxB, *UcB, *VhB;
  if (big_ws) {
    xbf = (unsigned short*)(ws + XC_BYTES);
    WxB = (unsigned short*)(ws + XC_BYTES + XBF_BYTES);
    UcB = (unsigned short*)(ws + XC_BYTES + XBF_BYTES + WX_BYTES);
    VhB = (unsigned short*)(ws + XC_BYTES + XBF_BYTES + WX_BYTES + UC_BYTES);
  } else {
    xbf = nullptr;
    WxB = nullptr;
    UcB = (unsigned short*)(ws + XC_BYTES);
    VhB = (unsigned short*)(ws + XC_BYTES + UC_BYTES);
  }

  // one-time fp32 -> bf16 weight (and optionally activation) conversion
  cvt_bf16<<<512,  256, 0, stream>>>(Uc, UcB, 1024 * 512);
  cvt_bf16<<<1024, 256, 0, stream>>>(Vh, VhB, 1024 * 1024);
  if (big_ws) {
    cvt_bf16<<<512,   256, 0, stream>>>(Wx, WxB, 512 * 1024);
    cvt_bf16<<<32768, 256, 0, stream>>>(x, xbf, 32768 * 1024);
    // async-staged bf16 GEMM: grid = 256 (M) x 4 (N); LDS = 4 x 10240 B
    gemm_xc_async<<<1024, 256, 4 * G_BUF, stream>>>(xbf, WxB, bx, xcb);
  } else {
    gemm_xc_cvt<<<1024, 256, 2 * 128 * G_PAD * 2, stream>>>(x, Wx, bx, xcb);
  }

  // persistent batch-partitioned scan: 4 blocks (16 batch rows each)
  size_t scan_lds = (size_t)16 * 512 * 4 + (size_t)16 * S_CP * 2 + (size_t)16 * S_HP * 2;
  scrn_scan<<<4, 256, scan_lds, stream>>>(xcb, h0, c0, UcB, VhB, out);
}
